// ResidualVectorQuantizer_37924561224278
// MI455X (gfx1250) — matrix operand order repaired and checked by reading, not compiled, not run on hardware
//
#include <hip/hip_runtime.h>

// ---------------------------------------------------------------------------
// Residual VQ forward, fused, WMMA bf16 distance GEMM (gfx1250 / CDNA5).
//   input:     (B=32, D=256, T=2048) f32
//   codebooks: (S=8, K=1024, D=256) f32
//   out: output (B,D,T) f32  ++  indices (B,S,T) as float(idx)
//
// v3: fix async builtin pointee type -> vector_size(16) int, matching the
//     builtin's declared parameter ('__device__ v4i*' global src,
//     '__shared__ v4i*' LDS dst).
// ---------------------------------------------------------------------------

typedef __attribute__((ext_vector_type(16))) __bf16 v16bf;
typedef __attribute__((ext_vector_type(8)))  float  v8f;
typedef int v4i __attribute__((vector_size(16)));   // matches builtin param

#define AS1 __attribute__((address_space(1)))
#define AS3 __attribute__((address_space(3)))

#if defined(__has_builtin)
#if __has_builtin(__builtin_amdgcn_global_load_async_to_lds_b128)
#define HAVE_ASYNC_LDS 1
#endif
#endif
#ifndef HAVE_ASYNC_LDS
#define HAVE_ASYNC_LDS 0
#endif

#define BB 32
#define DD 256
#define TT 2048
#define SS 8
#define KK 1024
#define NTOK (BB * TT)

#define WAVES 4
#define TILES 2                    // 16-row A tiles per wave -> M=32 per wave
#define WGTOK (WAVES * TILES * 16) // 128 tokens per workgroup
#define ROWSTRIDE 260              // residual LDS row stride (floats), bank-spread
#define BROW 264                   // B LDS row stride (bf16): 512B payload + 16B pad
#define BUFU16 (16 * BROW)         // one 16-code B buffer, in ushorts

__device__ __forceinline__ unsigned short f2bf(float f) {
  unsigned int u = __float_as_uint(f);
  unsigned int r = u + 0x7FFFu + ((u >> 16) & 1u);
  return (unsigned short)(r >> 16);
}

union BFrag {
  v16bf v;
  unsigned int w[8];
  uint4 q[2];
};

__device__ __forceinline__ void wait_async_all() {
#if HAVE_ASYNC_LDS
#if __has_builtin(__builtin_amdgcn_s_wait_asynccnt)
  __builtin_amdgcn_s_wait_asynccnt(0);
#else
  asm volatile("s_wait_asynccnt 0x0" ::: "memory");
#endif
#endif
}

// Copy one 16-code tile (16 rows x 512B payload) global -> LDS (row stride
// 528B). 512 x 16B chunks spread over 128 threads (4 each), coalesced.
__device__ __forceinline__ void copy_tile(unsigned short* __restrict__ dst,
                                          const char* __restrict__ gsrc,
                                          int code0, int tid) {
#pragma unroll
  for (int i = 0; i < 4; ++i) {
    const int c = i * 128 + tid;
    const int row = c >> 5;
    const int off = (c & 31) << 4;
    const char* g = gsrc + (size_t)(code0 + row) * 512 + off;
    char* l = (char*)dst + row * 528 + off;
#if HAVE_ASYNC_LDS
    __builtin_amdgcn_global_load_async_to_lds_b128((AS1 v4i*)g, (AS3 v4i*)l, 0, 0);
#else
    *(uint4*)l = *(const uint4*)g;
#endif
  }
}

// ---------------------------------------------------------------------------
// Prep: fp32 codebooks -> bf16 (workspace) + per-code squared norms.
// ---------------------------------------------------------------------------
__global__ __launch_bounds__(256) void rvq_prep(const float* __restrict__ cb,
                                                unsigned short* __restrict__ cb_bf,
                                                float* __restrict__ enorm) {
  const int lane = threadIdx.x & 31;
  const int wave = threadIdx.x >> 5;
  const int code = blockIdx.x * 8 + wave; // SS*KK = 8192 codes
  const float* row = cb + (size_t)code * DD;
  const int d0 = lane * 8;

  float4 a = *(const float4*)(row + d0);
  float4 b = *(const float4*)(row + d0 + 4);

  uint4 packed;
  packed.x = (unsigned int)f2bf(a.x) | ((unsigned int)f2bf(a.y) << 16);
  packed.y = (unsigned int)f2bf(a.z) | ((unsigned int)f2bf(a.w) << 16);
  packed.z = (unsigned int)f2bf(b.x) | ((unsigned int)f2bf(b.y) << 16);
  packed.w = (unsigned int)f2bf(b.z) | ((unsigned int)f2bf(b.w) << 16);
  *(uint4*)(cb_bf + (size_t)code * DD + d0) = packed;

  float ss = a.x * a.x + a.y * a.y + a.z * a.z + a.w * a.w +
             b.x * b.x + b.y * b.y + b.z * b.z + b.w * b.w;
#pragma unroll
  for (int off = 16; off >= 1; off >>= 1) ss += __shfl_xor(ss, off, 32);
  if (lane == 0) enorm[code] = ss;
}

// ---------------------------------------------------------------------------
// Main fused kernel.
// ---------------------------------------------------------------------------
__global__ __launch_bounds__(128) void rvq_main(const float* __restrict__ inp,
                                                const float* __restrict__ cb,
                                                const unsigned short* __restrict__ cb_bf,
                                                const float* __restrict__ enorm,
                                                float* __restrict__ out_q,
                                                float* __restrict__ out_idx) {
  __shared__ float sres[WGTOK * ROWSTRIDE];                     // 133,120 B
  __shared__ __align__(16) unsigned short sbuf[2 * BUFU16];     //  16,896 B
  __shared__ __align__(16) float senorm[KK];                    //   4,096 B

  const int tid  = threadIdx.x;
  const int lane = tid & 31;
  const int wave = tid >> 5;
  const int half = lane >> 4;
  const int l15  = lane & 15;

  const int wgTok0 = blockIdx.x * WGTOK;        // 128 | 2048 -> never crosses b
  const int b   = wgTok0 / TT;
  const int t0w = (wgTok0 - b * TT) + wave * (TILES * 16);
  const int rowbase = wave * (TILES * 16);      // this wave's rows in sres

  // ---- load 32 tokens x 256 dims into LDS, coalesced along T ----
  {
    const float* base = inp + ((size_t)b * DD) * TT + t0w;
#pragma unroll
    for (int g = 0; g < TILES; ++g) {
      for (int dp = 0; dp < DD; dp += 2) {
        const int d = dp + half;
        sres[(rowbase + g * 16 + l15) * ROWSTRIDE + d] =
            base[(size_t)d * TT + g * 16 + l15];
      }
    }
  }
  // rows are wave-private; first cross-wave barrier is in the stage prologue

  for (int s = 0; s < SS; ++s) {
    const char*  cbs = (const char*)(cb_bf + (size_t)s * KK * DD);
    const float* ens = enorm + s * KK;
    const float* cfs = cb + (size_t)s * KK * DD;

    // ---- bf16 A fragments (ISA 16x32 layout), 2 tiles x 8 K-chunks ----
    BFrag af[TILES][8];
    const int koff = half * 8;
#pragma unroll
    for (int t = 0; t < TILES; ++t) {
#pragma unroll
      for (int ck = 0; ck < 8; ++ck) {
        const float* rp =
            sres + (rowbase + t * 16 + l15) * ROWSTRIDE + ck * 32 + koff;
        float4 f0 = *(const float4*)(rp);
        float4 f1 = *(const float4*)(rp + 4);
        float4 f2 = *(const float4*)(rp + 16);
        float4 f3 = *(const float4*)(rp + 20);
        af[t][ck].w[0] = (unsigned int)f2bf(f0.x) | ((unsigned int)f2bf(f0.y) << 16);
        af[t][ck].w[1] = (unsigned int)f2bf(f0.z) | ((unsigned int)f2bf(f0.w) << 16);
        af[t][ck].w[2] = (unsigned int)f2bf(f1.x) | ((unsigned int)f2bf(f1.y) << 16);
        af[t][ck].w[3] = (unsigned int)f2bf(f1.z) | ((unsigned int)f2bf(f1.w) << 16);
        af[t][ck].w[4] = (unsigned int)f2bf(f2.x) | ((unsigned int)f2bf(f2.y) << 16);
        af[t][ck].w[5] = (unsigned int)f2bf(f2.z) | ((unsigned int)f2bf(f2.w) << 16);
        af[t][ck].w[6] = (unsigned int)f2bf(f3.x) | ((unsigned int)f2bf(f3.y) << 16);
        af[t][ck].w[7] = (unsigned int)f2bf(f3.z) | ((unsigned int)f2bf(f3.w) << 16);
      }
    }

    float bestv[TILES][8];
    int   besti[TILES][8];
#pragma unroll
    for (int t = 0; t < TILES; ++t)
#pragma unroll
      for (int v = 0; v < 8; ++v) { bestv[t][v] = 3.4e38f; besti[t][v] = 0; }

    // ---- stage prologue: stage tile 0 + ||e||^2 into LDS ----
    copy_tile(sbuf, cbs, 0, tid);
    {
      const float* ep = ens + tid * 8;
      float4 e0 = *(const float4*)(ep);
      float4 e1 = *(const float4*)(ep + 4);
      *(float4*)(senorm + tid * 8)     = e0;
      *(float4*)(senorm + tid * 8 + 4) = e1;
    }
    wait_async_all();
    __syncthreads();

    // ---- main loop: 64 N-tiles of 16 codes, double-buffered B in LDS ----
    for (int nt = 0; nt < KK / 16; ++nt) {
      if (nt + 1 < KK / 16)
        copy_tile(sbuf + ((nt + 1) & 1) * BUFU16, cbs, (nt + 1) * 16, tid);

      // B 32x16 layout: column n = l15; lanes 0-15 K 0..7,16..23; 16-31 K 8..15,24..31
      const unsigned short* bbase =
          sbuf + (nt & 1) * BUFU16 + l15 * BROW + half * 8;

      v8f c0 = {};
      v8f c1 = {};
#pragma unroll
      for (int ck = 0; ck < 8; ++ck) {
        BFrag bf;
        bf.q[0] = *(const uint4*)(bbase + ck * 32);
        bf.q[1] = *(const uint4*)(bbase + ck * 32 + 16);
        c0 = __builtin_amdgcn_wmma_f32_16x16x32_bf16(
                 false, af[0][ck].v, false, bf.v, (short)0, c0, false, false);
        c1 = __builtin_amdgcn_wmma_f32_16x16x32_bf16(
                 false, af[1][ck].v, false, bf.v, (short)0, c1, false, false);
      }

      const int   myn = nt * 16 + l15;
      const float en  = senorm[myn];
#pragma unroll
      for (int v = 0; v < 8; ++v) {
        const float s0 = en - 2.0f * c0[v];   // ||r||^2 constant per row: drop
        if (s0 < bestv[0][v]) { bestv[0][v] = s0; besti[0][v] = myn; }
        const float s1 = en - 2.0f * c1[v];
        if (s1 < bestv[1][v]) { bestv[1][v] = s1; besti[1][v] = myn; }
      }

      wait_async_all();   // next buffer landed
      __syncthreads();    // all waves done reading current buffer
    }

    // ---- per-row argmin across the 16 lanes of each half ----
#pragma unroll
    for (int off = 1; off < 16; off <<= 1) {
#pragma unroll
      for (int t = 0; t < TILES; ++t) {
#pragma unroll
        for (int v = 0; v < 8; ++v) {
          const float ov = __shfl_xor(bestv[t][v], off, 32);
          const int   oi = __shfl_xor(besti[t][v], off, 32);
          if (ov < bestv[t][v] || (ov == bestv[t][v] && oi < besti[t][v])) {
            bestv[t][v] = ov; besti[t][v] = oi;
          }
        }
      }
    }

    // ---- gather (exact fp32 codebook), residual update, index output ----
#pragma unroll
    for (int r = 0; r < TILES * 16; ++r) {
      const int t  = r >> 4;
      const int rr = r & 15;
      const int idx = __shfl(besti[t][rr & 7], (rr < 8) ? 0 : 16, 32);
      if (lane == 0) {
        out_idx[((size_t)b * SS + s) * TT + (t0w + r)] = (float)idx;
      }
      const float* q = cfs + (size_t)idx * DD + lane * 8;
      float4 q0 = *(const float4*)(q);
      float4 q1 = *(const float4*)(q + 4);
      float* rp = sres + (rowbase + r) * ROWSTRIDE + lane * 8;
      float4 r0 = *(const float4*)(rp);
      float4 r1 = *(const float4*)(rp + 4);
      r0.x -= q0.x; r0.y -= q0.y; r0.z -= q0.z; r0.w -= q0.w;
      r1.x -= q1.x; r1.y -= q1.y; r1.z -= q1.z; r1.w -= q1.w;
      *(float4*)(rp)     = r0;
      *(float4*)(rp + 4) = r1;
    }
  }

  // ---- output: quantized = x - final residual (sum of q's telescopes) ----
  {
    const float* base  = inp   + ((size_t)b * DD) * TT + t0w;
    float*       obase = out_q + ((size_t)b * DD) * TT + t0w;
#pragma unroll
    for (int g = 0; g < TILES; ++g) {
      for (int dp = 0; dp < DD; dp += 2) {
        const int d = dp + half;
        const float x = base[(size_t)d * TT + g * 16 + l15];
        obase[(size_t)d * TT + g * 16 + l15] =
            x - sres[(rowbase + g * 16 + l15) * ROWSTRIDE + d];
      }
    }
  }
}

// ---------------------------------------------------------------------------
extern "C" void kernel_launch(void* const* d_in, const int* in_sizes, int n_in,
                              void* d_out, int out_size, void* d_ws, size_t ws_size,
                              hipStream_t stream) {
  const float* inp = (const float*)d_in[0];   // (32,256,2048) f32
  const float* cb  = (const float*)d_in[1];   // (8,1024,256) f32

  unsigned short* cb_bf = (unsigned short*)d_ws;                          // 4 MB
  float* enorm = (float*)((char*)d_ws +
                          (size_t)SS * KK * DD * sizeof(unsigned short)); // 32 KB

  float* out_q   = (float*)d_out;                  // (B,D,T)
  float* out_idx = out_q + (size_t)BB * DD * TT;   // (B,S,T) as float(idx)

  rvq_prep<<<(SS * KK) / 8, 256, 0, stream>>>(cb, cb_bf, enorm);
  rvq_main<<<NTOK / WGTOK, 128, 0, stream>>>(inp, cb, cb_bf, enorm, out_q, out_idx);
}
